// DisentangleEncoder_9165460209683
// MI455X (gfx1250) — compile-verified, matrix operand order checked by reading, not run on hardware
//
#include <hip/hip_runtime.h>
#include <hip/hip_bf16.h>

#define BB 64
#define LL 200
#define DD 256
#define KK 8
#define ML (BB * LL)      /* 12800 rows per stream */
#define NROWS (2 * ML)    /* 25600 */
#define EPSF 1e-5f
#define SCALEF 0.0625f    /* 1/sqrt(256) */

typedef __attribute__((ext_vector_type(16))) __bf16 bf16x16;
typedef __attribute__((ext_vector_type(8)))  float  floatx8;
typedef __attribute__((ext_vector_type(4)))  unsigned int u32x4;
typedef __attribute__((ext_vector_type(8)))  int i32x8;
typedef __attribute__((ext_vector_type(4)))  int i32x4;

#if defined(__has_builtin)
#if __has_builtin(__builtin_amdgcn_tensor_load_to_lds) && __has_builtin(__builtin_amdgcn_s_wait_tensorcnt)
#define HAVE_TDM 1
#endif
#endif

union FragAB { bf16x16 v; uint4 q[2]; };
union Acc    { floatx8 v; float f[8]; };

__device__ __forceinline__ unsigned short f32_to_bf16(float f) {
    unsigned int u = __float_as_uint(f);
    u += 0x7FFFu + ((u >> 16) & 1u);       // round-to-nearest-even
    return (unsigned short)(u >> 16);
}
__device__ __forceinline__ float bf16_to_f32(unsigned short h) {
    return __uint_as_float(((unsigned int)h) << 16);
}

// ---------------- kernel 1: W_w -> bf16 ----------------
__global__ void k_cvt_w(const float* __restrict__ W, unsigned short* __restrict__ Wb) {
    int i = blockIdx.x * 256 + threadIdx.x;
    Wb[i] = f32_to_bf16(W[i]);
}

// ---------------- kernel 2: ln2(intentions) ----------------
__global__ void k_ln_intent(const float* __restrict__ intent,
                            const float* __restrict__ g2, const float* __restrict__ b2,
                            float* __restrict__ ln2) {
    __shared__ float sA[256], sB[256];
    int t = threadIdx.x;
    for (int k = 0; k < KK; k++) {
        float x = intent[k * DD + t];
        sA[t] = x; sB[t] = x * x; __syncthreads();
        for (int s = 128; s > 0; s >>= 1) {
            if (t < s) { sA[t] += sA[t + s]; sB[t] += sB[t + s]; }
            __syncthreads();
        }
        float mu  = sA[0] * (1.0f / DD);
        float var = sB[0] * (1.0f / DD) - mu * mu;
        ln2[k * DD + t] = (x - mu) * rsqrtf(var + EPSF) * g2[t] + b2[t];
        __syncthreads();
    }
}

// ---------------- kernel 3: per-row prep ----------------
__global__ void __launch_bounds__(256) k_prep_rows(
        const float* __restrict__ localE, const float* __restrict__ globalE,
        const float* __restrict__ pos,    const float* __restrict__ ln2,
        const float* __restrict__ g1, const float* __restrict__ b1,
        const float* __restrict__ g4, const float* __restrict__ b4,
        unsigned short* __restrict__ khb,
        float* __restrict__ muO, float* __restrict__ varO,
        float* __restrict__ scoreO) {
    __shared__ float sA[256], sB[256], s1[256], sLog[KK];
    int t = threadIdx.x;
    int idx = blockIdx.x;                 // 0..NROWS-1
    int disen = idx / ML;
    int r = idx - disen * ML;
    int l = r % LL;
    const float* src = disen ? globalE : localE;
    float x = src[(size_t)r * DD + t];

    // stats of raw row (for ln5 algebra)
    sA[t] = x; sB[t] = x * x; __syncthreads();
    for (int s = 128; s > 0; s >>= 1) {
        if (t < s) { sA[t] += sA[t + s]; sB[t] += sB[t + s]; }
        __syncthreads();
    }
    float mu  = sA[0] * (1.0f / DD);
    float var = sB[0] * (1.0f / DD) - mu * mu;
    if (t == 0) { muO[idx] = mu; varO[idx] = var; }
    float v1 = (x - mu) * rsqrtf(var + EPSF) * g1[t] + b1[t];   // ln1 row
    s1[t] = v1;
    __syncthreads();

    // key_hat = ln4(x + pos)
    float xp = x + pos[l * DD + t];
    sA[t] = xp; sB[t] = xp * xp; __syncthreads();
    for (int s = 128; s > 0; s >>= 1) {
        if (t < s) { sA[t] += sA[t + s]; sB[t] += sB[t + s]; }
        __syncthreads();
    }
    float mu4  = sA[0] * (1.0f / DD);
    float var4 = sB[0] * (1.0f / DD) - mu4 * mu4;
    float kh = (xp - mu4) * rsqrtf(var4 + EPSF) * g4[t] + b4[t];
    khb[(size_t)idx * DD + t] = f32_to_bf16(kh);

    // scores: wave k computes dot(ln1_row, ln2_k), then K-softmax
    int wv = t >> 5, lane = t & 31;
    float p = 0.f;
    for (int j = 0; j < 8; j++) {
        int d = lane + 32 * j;
        p += s1[d] * ln2[wv * DD + d];
    }
    for (int m = 16; m > 0; m >>= 1) p += __shfl_xor(p, m, 32);
    if (lane == 0) sLog[wv] = p * SCALEF;
    __syncthreads();
    if (t == 0) {
        float mx = -1e30f;
        for (int k = 0; k < KK; k++) mx = fmaxf(mx, sLog[k]);
        float e[KK], sum = 0.f;
        for (int k = 0; k < KK; k++) { e[k] = __expf(sLog[k] - mx); sum += e[k]; }
        float inv = 1.f / sum;
        for (int k = 0; k < KK; k++) scoreO[(size_t)idx * KK + k] = e[k] * inv;
    }
}

// ---------------- kernel 4: q = ln3(x[seq-1] + pos + rou) ----------------
__global__ void k_prep_q(const float* __restrict__ localE, const float* __restrict__ globalE,
                         const float* __restrict__ pos, const float* __restrict__ rou,
                         const float* __restrict__ g3, const float* __restrict__ b3,
                         const int* __restrict__ seq_len, float* __restrict__ qO) {
    __shared__ float sA[256], sB[256];
    int t = threadIdx.x;
    int idx = blockIdx.x;                 // disen*64 + b
    int disen = idx >> 6;
    int b = idx & 63;
    const float* src = disen ? globalE : localE;
    int li = seq_len[b] - 1;
    float x = src[((size_t)b * LL + li) * DD + t] + pos[(size_t)li * DD + t] + rou[t];
    sA[t] = x; sB[t] = x * x; __syncthreads();
    for (int s = 128; s > 0; s >>= 1) {
        if (t < s) { sA[t] += sA[t + s]; sB[t] += sB[t + s]; }
        __syncthreads();
    }
    float mu  = sA[0] * (1.0f / DD);
    float var = sB[0] * (1.0f / DD) - mu * mu;
    qO[(size_t)idx * DD + t] = (x - mu) * rsqrtf(var + EPSF) * g3[t] + b3[t];
}

// ---------------- kernel 5: WMMA GEMM keyv = kh + relu(kh @ W^T + b) ----------------
// Block = 256 thr = 8 waves covering 16 rows x 256 cols. A tile (16x256 bf16, 8KB)
// is DMA'd into LDS once per block via the Tensor Data Mover; waves read A frags
// from LDS (ds_load_b128) and B frags (W, L2-resident) from global.
__global__ void __launch_bounds__(256) k_gemm_keyv(
        const unsigned short* __restrict__ khb,
        const unsigned short* __restrict__ Wb,
        const float* __restrict__ Wbias,
        float* __restrict__ keyv) {
    __shared__ __align__(16) unsigned short shA[16 * DD];   // 8 KB A tile
    int lane = threadIdx.x & 31;
    int wv   = threadIdx.x >> 5;          // 0..7 -> N tiles [wv*32, wv*32+32)
    int rowBase = blockIdx.x * 16;

#ifdef HAVE_TDM
    if (threadIdx.x < 32) {
        unsigned lds_addr = (unsigned)(size_t)(&shA[0]);
        unsigned long long ga = (unsigned long long)(size_t)(khb + (size_t)rowBase * DD);
        // D# group 0: count=1 | lds_addr | global_addr | type=2
        u32x4 g0 = { 1u, lds_addr, (unsigned)ga,
                     (unsigned)(ga >> 32) | 0x80000000u };
        // D# group 1: data_size=2B; tensor_dim0=256; tensor_dim1=NROWS;
        //             tile_dim0=256, tile_dim1=16; tensor_dim0_stride=256
        i32x8 g1 = { (int)0x00010000u,
                     (int)(256u << 16),
                     (int)((unsigned)NROWS << 16),
                     (int)(256u << 16),
                     16, 256, 0, 0 };
        i32x4 g2 = { 0, 0, 0, 0 };
        i32x4 g3 = { 0, 0, 0, 0 };
        i32x8 g4 = { 0, 0, 0, 0, 0, 0, 0, 0 };
        __builtin_amdgcn_tensor_load_to_lds(g0, g1, g2, g3, g4, 0);
        __builtin_amdgcn_s_wait_tensorcnt(0);
    }
#else
    {
        const uint4* srcq = (const uint4*)(khb + (size_t)rowBase * DD);
        uint4* dstq = (uint4*)shA;
        for (int i = threadIdx.x; i < (16 * DD) / 8; i += 256) dstq[i] = srcq[i];
    }
#endif
    __syncthreads();

    int m    = lane & 15;
    int koff = (lane >> 4) * 8;           // 16-bit A/B lane layout K-offset
    const unsigned short* arow  = &shA[m * DD];
    int n0 = wv * 32 + (lane & 15);
    int n1 = n0 + 16;
    const unsigned short* b0row = Wb + (size_t)n0 * DD;  // B[k][n] = W[n][k]
    const unsigned short* b1row = Wb + (size_t)n1 * DD;

    Acc acc0, acc1;
    for (int i = 0; i < 8; i++) { acc0.f[i] = 0.f; acc1.f[i] = 0.f; }

    for (int k0 = 0; k0 < DD; k0 += 32) {
        __builtin_prefetch((const void*)(b0row + k0 + 64), 0, 1);
        FragAB a, fb0, fb1;
        a.q[0]   = *reinterpret_cast<const uint4*>(arow  + k0 + koff);
        a.q[1]   = *reinterpret_cast<const uint4*>(arow  + k0 + 16 + koff);
        fb0.q[0] = *reinterpret_cast<const uint4*>(b0row + k0 + koff);
        fb0.q[1] = *reinterpret_cast<const uint4*>(b0row + k0 + 16 + koff);
        fb1.q[0] = *reinterpret_cast<const uint4*>(b1row + k0 + koff);
        fb1.q[1] = *reinterpret_cast<const uint4*>(b1row + k0 + 16 + koff);
        acc0.v = __builtin_amdgcn_wmma_f32_16x16x32_bf16(
                     false, a.v, false, fb0.v, (short)0, acc0.v, false, false);
        acc1.v = __builtin_amdgcn_wmma_f32_16x16x32_bf16(
                     false, a.v, false, fb1.v, (short)0, acc1.v, false, false);
    }

    // epilogue: D layout -> lane col = lane&15, reg r row = r + (lane>>4)*8
    int colHalf = lane & 15;
    int mOff    = (lane >> 4) * 8;
    int c0 = wv * 32 + colHalf;
    int c1 = c0 + 16;
    float bias0 = Wbias[c0];
    float bias1 = Wbias[c1];
    for (int r = 0; r < 8; r++) {
        int row = rowBase + r + mOff;
        size_t base = (size_t)row * DD;
        float kh0 = bf16_to_f32(shA[(r + mOff) * DD + c0]);
        float kh1 = bf16_to_f32(shA[(r + mOff) * DD + c1]);
        keyv[base + c0] = kh0 + fmaxf(acc0.f[r] + bias0, 0.f);
        keyv[base + c1] = kh1 + fmaxf(acc1.f[r] + bias1, 0.f);
    }
}

// ---------------- kernel 6: attention softmax over L ----------------
__global__ void __launch_bounds__(256) k_attn(
        const float* __restrict__ qO, const float* __restrict__ keyv,
        float* __restrict__ wO) {
    __shared__ __align__(16) float qs[DD];
    __shared__ float lg[256];
    __shared__ float red[256];
    int t = threadIdx.x;
    int idx = blockIdx.x;                  // disen*64 + b
    qs[t] = qO[(size_t)idx * DD + t];
    lg[t] = -1e30f;
    __syncthreads();
    int wv = t >> 5, lane = t & 31;
    float4 qa = *(const float4*)(qs + lane * 8);
    float4 qb = *(const float4*)(qs + lane * 8 + 4);
    for (int i = 0; i < 25; i++) {         // 8 waves * 25 rows = 200
        int l = wv * 25 + i;
        const float* kr = keyv + ((size_t)idx * LL + l) * DD + lane * 8;
        float4 ka = *(const float4*)(kr);
        float4 kb = *(const float4*)(kr + 4);
        float p = qa.x * ka.x + qa.y * ka.y + qa.z * ka.z + qa.w * ka.w
                + qb.x * kb.x + qb.y * kb.y + qb.z * kb.z + qb.w * kb.w;
        for (int m2 = 16; m2 > 0; m2 >>= 1) p += __shfl_xor(p, m2, 32);
        if (lane == 0) lg[l] = p * SCALEF;
    }
    __syncthreads();
    red[t] = lg[t]; __syncthreads();
    for (int s = 128; s > 0; s >>= 1) {
        if (t < s) red[t] = fmaxf(red[t], red[t + s]);
        __syncthreads();
    }
    float mx = red[0]; __syncthreads();
    float e = (t < LL) ? __expf(lg[t] - mx) : 0.f;
    red[t] = e; __syncthreads();
    for (int s = 128; s > 0; s >>= 1) {
        if (t < s) red[t] += red[t + s];
        __syncthreads();
    }
    float inv = 1.f / red[0];
    if (t < LL) wO[(size_t)idx * LL + t] = e * inv;
}

// ---------------- kernel 7: fused output (b128 IO) ----------------
// out[b,k,l,:] = (xl-mu0)*rs0*g5 + (xg-mu1)*rs1*g5 + 2*b5,  rs = rsqrt(var + eps/c^2)
// Thread -> 4 contiguous d for 2 of the 8 K-slices (256 thr = 4 k-groups x 64 d-quads).
__global__ void __launch_bounds__(256) k_out(
        const float* __restrict__ localE, const float* __restrict__ globalE,
        const float* __restrict__ muO, const float* __restrict__ varO,
        const float* __restrict__ scoreO, const float* __restrict__ wO,
        const float* __restrict__ g5, const float* __restrict__ b5,
        float* __restrict__ out) {
    int t = threadIdx.x;
    int r = blockIdx.x;                    // b*LL + l
    int b = r / LL;
    int l = r - b * LL;
    int kq = t >> 6;                       // 0..3
    int dq = (t & 63) * 4;                 // 0..252
    float4 xl4 = *(const float4*)(localE  + (size_t)r * DD + dq);
    float4 xg4 = *(const float4*)(globalE + (size_t)r * DD + dq);
    float4 gg  = *(const float4*)(g5 + dq);
    float4 bb4 = *(const float4*)(b5 + dq);
    float mu0 = muO[r],      var0 = varO[r];
    float mu1 = muO[ML + r], var1 = varO[ML + r];
    float w0 = wO[(size_t)b * LL + l];
    float w1 = wO[(size_t)(64 + b) * LL + l];
    float d0x = (xl4.x - mu0) * gg.x, d1x = (xg4.x - mu1) * gg.x;
    float d0y = (xl4.y - mu0) * gg.y, d1y = (xg4.y - mu1) * gg.y;
    float d0z = (xl4.z - mu0) * gg.z, d1z = (xg4.z - mu1) * gg.z;
    float d0w = (xl4.w - mu0) * gg.w, d1w = (xg4.w - mu1) * gg.w;
    for (int kk = 0; kk < 2; kk++) {
        int k = kq + 4 * kk;
        float c0 = scoreO[(size_t)r * KK + k] * w0;
        float c1 = scoreO[(size_t)(ML + r) * KK + k] * w1;
        float rs0 = rsqrtf(var0 + EPSF / (c0 * c0));
        float rs1 = rsqrtf(var1 + EPSF / (c1 * c1));
        float4 o;
        o.x = d0x * rs0 + d1x * rs1 + 2.f * bb4.x;
        o.y = d0y * rs0 + d1y * rs1 + 2.f * bb4.y;
        o.z = d0z * rs0 + d1z * rs1 + 2.f * bb4.z;
        o.w = d0w * rs0 + d1w * rs1 + 2.f * bb4.w;
        *(float4*)(out + (((size_t)b * KK + k) * LL + l) * DD + dq) = o;
    }
}

extern "C" void kernel_launch(void* const* d_in, const int* in_sizes, int n_in,
                              void* d_out, int out_size, void* d_ws, size_t ws_size,
                              hipStream_t stream) {
    (void)in_sizes; (void)n_in; (void)out_size; (void)ws_size;
    const float* localE  = (const float*)d_in[0];
    const float* globalE = (const float*)d_in[1];
    const float* intent  = (const float*)d_in[2];
    const float* pos     = (const float*)d_in[3];
    const float* rou     = (const float*)d_in[4];
    const float* W_w     = (const float*)d_in[5];
    const float* W_b     = (const float*)d_in[6];
    const float* g1 = (const float*)d_in[7],  *b1 = (const float*)d_in[8];
    const float* g2 = (const float*)d_in[9],  *b2 = (const float*)d_in[10];
    const float* g3 = (const float*)d_in[11], *b3 = (const float*)d_in[12];
    const float* g4 = (const float*)d_in[13], *b4 = (const float*)d_in[14];
    const float* g5 = (const float*)d_in[15], *b5 = (const float*)d_in[16];
    const int* seq_len = (const int*)d_in[17];
    float* out = (float*)d_out;

    char* p = (char*)d_ws;
    unsigned short* khb = (unsigned short*)p; p += (size_t)NROWS * DD * 2;  // 13.1 MB
    unsigned short* Wbf = (unsigned short*)p; p += (size_t)DD * DD * 2;     // 128 KB
    float* ln2    = (float*)p; p += (size_t)KK * DD * 4;
    float* muO    = (float*)p; p += (size_t)NROWS * 4;
    float* varO   = (float*)p; p += (size_t)NROWS * 4;
    float* scoreO = (float*)p; p += (size_t)NROWS * KK * 4;
    float* qO     = (float*)p; p += (size_t)2 * BB * DD * 4;
    float* wO     = (float*)p; p += (size_t)2 * BB * LL * 4;
    float* keyv   = (float*)p; p += (size_t)NROWS * DD * 4;                 // 26.2 MB

    k_cvt_w<<<(DD * DD) / 256, 256, 0, stream>>>(W_w, Wbf);
    k_ln_intent<<<1, 256, 0, stream>>>(intent, g2, b2, ln2);
    k_prep_rows<<<NROWS, 256, 0, stream>>>(localE, globalE, pos, ln2,
                                           g1, b1, g4, b4, khb, muO, varO, scoreO);
    k_prep_q<<<2 * BB, 256, 0, stream>>>(localE, globalE, pos, rou, g3, b3, seq_len, qO);
    k_gemm_keyv<<<NROWS / 16, 256, 0, stream>>>(khb, Wbf, W_b, keyv);
    k_attn<<<2 * BB, 256, 0, stream>>>(qO, keyv, wO);
    k_out<<<ML, 256, 0, stream>>>(localE, globalE, muO, varO, scoreO, wO, g5, b5, out);
}
